// GraphSAGE_12146167513369
// MI455X (gfx1250) — matrix operand order repaired and checked by reading, not compiled, not run on hardware
//
#include <hip/hip_runtime.h>
#include <hip/hip_bf16.h>

#define N_NODES 100000
#define N_EDGES 1600000
#define FEAT 128
#define HID 128

typedef __attribute__((ext_vector_type(16))) __bf16 v16bf;
typedef __attribute__((ext_vector_type(8)))  float  v8f;

union BfFrag {
    v16bf          bf;
    unsigned short u16[16];
    unsigned int   u32[8];
};

// Pack two floats into one dword of bf16x2.
// Preferred: HW packed converter if the builtin exists.
// Fallback: round-to-nearest (ties away): bias each f32 by 0x8000, then a
// single v_perm_b32 splices the two high halves -> 3 VALU ops per pair.
__device__ __forceinline__ unsigned int pack_bf16(float x, float y) {
#if __has_builtin(__builtin_amdgcn_cvt_pk_bf16_f32)
    auto r = __builtin_amdgcn_cvt_pk_bf16_f32(x, y);
    unsigned int u;
    __builtin_memcpy(&u, &r, 4);
    return u;
#else
    unsigned int ux = __float_as_uint(x) + 0x8000u;
    unsigned int uy = __float_as_uint(y) + 0x8000u;
    // D.byte0 = x.byte2 (idx 2), D.byte1 = x.byte3 (idx 3),
    // D.byte2 = y.byte2 (idx 6), D.byte3 = y.byte3 (idx 7); src1=x, src0=y
    return __builtin_amdgcn_perm(uy, ux, 0x07060302u);
#endif
}

// ---------------------------------------------------------------- utilities
__global__ void zero_kernel(float* __restrict__ p, long long n) {
    long long i = (long long)blockIdx.x * blockDim.x + threadIdx.x;
    long long stride = (long long)gridDim.x * blockDim.x;
    for (; i < n; i += stride) p[i] = 0.0f;
}

__global__ void cvt_w_kernel(const float* __restrict__ w,
                             unsigned int* __restrict__ o, int nPairs) {
    int i = blockIdx.x * blockDim.x + threadIdx.x;
    if (i < nPairs) {
        float2 v = *(const float2*)(w + 2 * i);
        o[i] = pack_bf16(v.x, v.y);
    }
}

// --------------------------------------------------------------- scatter
// One wave (32 lanes) per edge; each lane moves 4 consecutive features.
// Feature matrix (51MB) + agg buffer fit in the 192MB L2, so the random
// gather and the f32 atomics resolve at L2 speed, not HBM.
__global__ void scatter_kernel(const float* __restrict__ feat,
                               const long long* __restrict__ srcIdx,
                               const long long* __restrict__ dstIdx,
                               float* __restrict__ agg,
                               float* __restrict__ deg,
                               int nEdges) {
    long long idx = (long long)blockIdx.x * blockDim.x + threadIdx.x;
    long long total = (long long)nEdges * 32;
    if (idx >= total) return;
    int e    = (int)(idx >> 5);
    int lane = (int)(idx & 31);
    int s = (int)srcIdx[e];
    int d = (int)dstIdx[e];
    const float4 v = *(const float4*)(feat + (size_t)s * FEAT + lane * 4);
    float* a = agg + (size_t)d * FEAT + lane * 4;
    atomicAdd(a + 0, v.x);
    atomicAdd(a + 1, v.y);
    atomicAdd(a + 2, v.z);
    atomicAdd(a + 3, v.w);
    if (deg != nullptr && lane == 0) atomicAdd(deg + d, 1.0f);
}

// A-fragment K index for 16-bit A 16x32 (ISA 7.12.2): lane half h, vgpr j
__device__ __forceinline__ int a_k_of(int j, int h) {
    return ((j >> 2) * 16) + h * 8 + (j & 3) * 2;
}

// ------------------------------------------------------------ layer-1 GEMM
// h1 = relu( (agg/deg) @ W1l^T + b1 + x @ W1r^T ), one wave per 16-node tile,
// full 128-wide output per wave: 8 N-tiles x 4 K-steps x 2 matrices = 64 WMMA.
__global__ __launch_bounds__(128) void gemm1_wmma(
    const float* __restrict__ agg, const float* __restrict__ deg,
    const float* __restrict__ x,
    const unsigned short* __restrict__ w1l_bf,
    const unsigned short* __restrict__ w1r_bf,
    const float* __restrict__ b1, float* __restrict__ h1) {

    const int lane = threadIdx.x & 31;
    const int wave = threadIdx.x >> 5;
    const int nodeBase = blockIdx.x * 64 + wave * 16;
    const int m = lane & 15;      // A row within tile / B,D column
    const int h = lane >> 4;      // lane half

    int node  = nodeBase + m;
    int nodeC = node < N_NODES ? node : N_NODES - 1;   // clamp loads, keep EXEC full
    const float invd = 1.0f / fmaxf(deg[nodeC], 1.0f);
    const float* aggRow = agg + (size_t)nodeC * FEAT;
    const float* xRow   = x   + (size_t)nodeC * FEAT;

    v8f acc[8];
    #pragma unroll
    for (int nt = 0; nt < 8; ++nt) acc[nt] = (v8f){0,0,0,0,0,0,0,0};

    #pragma unroll
    for (int kt = 0; kt < 4; ++kt) {
        BfFrag aA, aX;
        #pragma unroll
        for (int j = 0; j < 8; ++j) {
            int k0 = kt * 32 + a_k_of(j, h);
            float2 va = *(const float2*)(aggRow + k0);
            float2 vx = *(const float2*)(xRow + k0);
            aA.u32[j] = pack_bf16(va.x * invd, va.y * invd);
            aX.u32[j] = pack_bf16(vx.x, vx.y);
        }
        #pragma unroll
        for (int nt = 0; nt < 8; ++nt) {
            int col = nt * 16 + m;            // B column = lane%16
            BfFrag bL, bR;
            #pragma unroll
            for (int j = 0; j < 8; ++j) {
                int k0 = kt * 32 + a_k_of(j, h);     // B: same K order as A fragment
                bL.u32[j] = *(const unsigned int*)(w1l_bf + (size_t)col * FEAT + k0);
                bR.u32[j] = *(const unsigned int*)(w1r_bf + (size_t)col * FEAT + k0);
            }
            acc[nt] = __builtin_amdgcn_wmma_f32_16x16x32_bf16(
                false, aA.bf, false, bL.bf, (short)0, acc[nt], false, false);
            acc[nt] = __builtin_amdgcn_wmma_f32_16x16x32_bf16(
                false, aX.bf, false, bR.bf, (short)0, acc[nt], false, false);
        }
    }

    // D layout: lane(h,n) vgpr v -> row v+8h, col nt*16+n
    #pragma unroll
    for (int nt = 0; nt < 8; ++nt) {
        int col = nt * 16 + m;
        float bias = b1[col];
        #pragma unroll
        for (int v = 0; v < 8; ++v) {
            int row = nodeBase + v + 8 * h;
            if (row < N_NODES) {
                float val = acc[nt][v] + bias;
                h1[(size_t)row * HID + col] = fmaxf(val, 0.0f);
            }
        }
    }
}

// ---------------------------------------------------- layer-2 GEMM + linear
// h2 = (agg2/deg) @ W2l^T + b2 + h1 @ W2r^T  (only 4 valid output columns),
// then out = h2 @ Wlin^T + blin via shfl_xor reduction over columns 0..3.
__global__ __launch_bounds__(128) void layer2_wmma(
    const float* __restrict__ agg, const float* __restrict__ deg,
    const float* __restrict__ h1,
    const float* __restrict__ W2l, const float* __restrict__ b2,
    const float* __restrict__ W2r, const float* __restrict__ Wlin,
    const float* __restrict__ blin, float* __restrict__ out) {

    const int lane = threadIdx.x & 31;
    const int wave = threadIdx.x >> 5;
    const int nodeBase = blockIdx.x * 64 + wave * 16;
    const int m = lane & 15;
    const int h = lane >> 4;

    int node  = nodeBase + m;
    int nodeC = node < N_NODES ? node : N_NODES - 1;
    const float invd = 1.0f / fmaxf(deg[nodeC], 1.0f);
    const float* aggRow = agg + (size_t)nodeC * HID;
    const float* hRow   = h1  + (size_t)nodeC * HID;

    v8f acc = (v8f){0,0,0,0,0,0,0,0};

    #pragma unroll
    for (int kt = 0; kt < 4; ++kt) {
        BfFrag aA, aH, bL, bR;
        #pragma unroll
        for (int j = 0; j < 8; ++j) {
            int k0 = kt * 32 + a_k_of(j, h);
            float2 va = *(const float2*)(aggRow + k0);
            float2 vh = *(const float2*)(hRow + k0);
            aA.u32[j] = pack_bf16(va.x * invd, va.y * invd);
            aH.u32[j] = pack_bf16(vh.x, vh.y);
        }
        #pragma unroll
        for (int j = 0; j < 8; ++j) {
            int k0 = kt * 32 + a_k_of(j, h);
            if (m < 4) {   // only 4 real output columns; rest zero
                float2 wl = *(const float2*)(W2l + (size_t)m * HID + k0);
                float2 wr = *(const float2*)(W2r + (size_t)m * HID + k0);
                bL.u32[j] = pack_bf16(wl.x, wl.y);
                bR.u32[j] = pack_bf16(wr.x, wr.y);
            } else {
                bL.u32[j] = 0u;
                bR.u32[j] = 0u;
            }
        }
        acc = __builtin_amdgcn_wmma_f32_16x16x32_bf16(
            false, aA.bf, false, bL.bf, (short)0, acc, false, false);
        acc = __builtin_amdgcn_wmma_f32_16x16x32_bf16(
            false, aH.bf, false, bR.bf, (short)0, acc, false, false);
    }

    // Finalize: row v+8h lives across lanes (columns); fold 4 classes -> 2 outputs.
    const float w0 = (m < 4) ? Wlin[m]     : 0.0f;   // Wlin[0][m]
    const float w1 = (m < 4) ? Wlin[4 + m] : 0.0f;   // Wlin[1][m]
    const float bb = (m < 4) ? b2[m]       : 0.0f;
    #pragma unroll
    for (int v = 0; v < 8; ++v) {
        float val = acc[v] + bb;          // columns >=4 accumulated 0, bias 0
        float p0 = w0 * val;
        float p1 = w1 * val;
        p0 += __shfl_xor(p0, 1, 32);
        p0 += __shfl_xor(p0, 2, 32);
        p1 += __shfl_xor(p1, 1, 32);
        p1 += __shfl_xor(p1, 2, 32);
        int row = nodeBase + v + 8 * h;
        if (m == 0 && row < N_NODES) {    // lanes 0 and 16 own the sums
            out[(size_t)row * 2 + 0] = p0 + blin[0];
            out[(size_t)row * 2 + 1] = p1 + blin[1];
        }
    }
}

// ------------------------------------------------------------------ launch
extern "C" void kernel_launch(void* const* d_in, const int* in_sizes, int n_in,
                              void* d_out, int out_size, void* d_ws, size_t ws_size,
                              hipStream_t stream) {
    const float*     x    = (const float*)d_in[0];
    const long long* ei   = (const long long*)d_in[1];   // int64 [2, E]
    const float*     W1l  = (const float*)d_in[2];
    const float*     b1   = (const float*)d_in[3];
    const float*     W1r  = (const float*)d_in[4];
    const float*     W2l  = (const float*)d_in[5];
    const float*     b2   = (const float*)d_in[6];
    const float*     W2r  = (const float*)d_in[7];
    const float*     Wlin = (const float*)d_in[8];
    const float*     blin = (const float*)d_in[9];
    float*           out  = (float*)d_out;

    // workspace layout: agg | deg | h1 | W1l_bf | W1r_bf  (~103 MB)
    char* ws = (char*)d_ws;
    float* agg = (float*)ws;                                             // N*128 f32
    float* deg = (float*)(ws + (size_t)N_NODES * FEAT * 4);              // N f32
    float* h1  = (float*)(ws + (size_t)N_NODES * (FEAT + 1) * 4);        // N*128 f32
    unsigned short* w1l_bf =
        (unsigned short*)(ws + (size_t)N_NODES * (2 * FEAT + 1) * 4);
    unsigned short* w1r_bf = w1l_bf + FEAT * HID;

    const long long* srcIdx = ei;
    const long long* dstIdx = ei + N_EDGES;

    const int scatterBlocks = (int)(((long long)N_EDGES * 32 + 255) / 256);
    const int gemmBlocks    = (N_NODES + 63) / 64;
    const int wPairs        = FEAT * HID / 2;

    // 1) zero agg + deg (contiguous)
    zero_kernel<<<2048, 256, 0, stream>>>(agg, (long long)N_NODES * FEAT + N_NODES);
    // 2) bf16 weight copies for layer 1
    cvt_w_kernel<<<(wPairs + 255) / 256, 256, 0, stream>>>(W1l, (unsigned int*)w1l_bf, wPairs);
    cvt_w_kernel<<<(wPairs + 255) / 256, 256, 0, stream>>>(W1r, (unsigned int*)w1r_bf, wPairs);
    // 3) scatter layer 1 (also builds degree)
    scatter_kernel<<<scatterBlocks, 256, 0, stream>>>(x, srcIdx, dstIdx, agg, deg, N_EDGES);
    // 4) layer-1 fused GEMM (WMMA bf16, f32 accum) + bias + relu
    gemm1_wmma<<<gemmBlocks, 128, 0, stream>>>(agg, deg, x, w1l_bf, w1r_bf, b1, h1);
    // 5) reuse agg buffer for layer 2 (deg is identical, keep it)
    zero_kernel<<<2048, 256, 0, stream>>>(agg, (long long)N_NODES * FEAT);
    // 6) scatter layer 2
    scatter_kernel<<<scatterBlocks, 256, 0, stream>>>(h1, srcIdx, dstIdx, agg, nullptr, N_EDGES);
    // 7) layer-2 GEMM (WMMA) + final linear, writes [N,2]
    layer2_wmma<<<gemmBlocks, 128, 0, stream>>>(agg, deg, h1, W2l, b2, W2r, Wlin, blin, out);
}
